// BatchedStarNetAttentionBlock_68951404970610
// MI455X (gfx1250) — matrix operand order repaired and checked by reading, not compiled, not run on hardware
//
#include <hip/hip_runtime.h>
#include <math.h>

// ---------------------------------------------------------------------------
// BatchedStarNetAttentionBlock for MI455X (gfx1250, wave32, WMMA)
// N=4096 tokens, D=256, H=8 heads, DH=32, B=32 batches of 128 tokens.
// ordering is guaranteed repeat(arange(32), 128) -> block-diagonal attention
// == 32 independent dense 128x128 attention problems per head.
// All matmuls through v_wmma_f32_16x16x32_bf16 (f32 accum); all LDS staging
// through GLOBAL_LOAD_ASYNC_TO_LDS_B128 (ASYNCcnt) when available; weights
// pre-transposed to n-major so every stage is 16-byte chunks.
// ---------------------------------------------------------------------------

typedef __attribute__((ext_vector_type(16))) __bf16 v16bf;
typedef __attribute__((ext_vector_type(8)))  float  v8f;
typedef int v4i_vs __attribute__((vector_size(16)));   // matches builtin param

#define NTOK   4096
#define DIM    256
#define NHEAD  8
#define DHEAD  32
#define NBATCH 32
#define SCALE_QK 0.17677669529663687f  // 1/sqrt(32)

// ---------------- async global -> LDS copy (CDNA5 ASYNCcnt path) -----------
#if __has_builtin(__builtin_amdgcn_global_load_async_to_lds_b128)
#define HAVE_ASYNC_LDS 1
#else
#define HAVE_ASYNC_LDS 0
#endif

__device__ __forceinline__ void cp16(void* lds, const void* g) {
#if HAVE_ASYNC_LDS
  __builtin_amdgcn_global_load_async_to_lds_b128(
      (__attribute__((address_space(1))) v4i_vs*)g,
      (__attribute__((address_space(3))) v4i_vs*)lds, 0, 0);
#else
  *(uint4*)lds = *(const uint4*)g;
#endif
}

__device__ __forceinline__ void async_wait_all() {
#if HAVE_ASYNC_LDS
#if __has_builtin(__builtin_amdgcn_s_wait_asynccnt)
  __builtin_amdgcn_s_wait_asynccnt(0);
#else
  asm volatile("s_wait_asynccnt 0x0" ::: "memory");
#endif
#endif
}

// ------------------------------ fragments ----------------------------------
union FragU { v16bf v; uint4 u[2]; };

// A-fragment (16x32 bf16, row-major source, stride in elements).
// lane m=L&15; elems 0..7 = K ks..ks+7, elems 8..15 = K ks+16..ks+23, ks=8*(L>>4).
__device__ __forceinline__ v16bf frag_a_rowmajor(const __bf16* p, int stride, int lane) {
  const __bf16* row = p + (lane & 15) * stride + ((lane >> 4) << 3);
  FragU f;
  f.u[0] = *(const uint4*)(row);
  f.u[1] = *(const uint4*)(row + 16);
  return f.v;
}

// B-fragment (32x16 bf16) from n-major tile Bt[n][k] (stride in elements).
// lane n=L&15; elems 0..7 = K ks..ks+7, elems 8..15 = K ks+16..ks+23, ks=8*(L>>4).
__device__ __forceinline__ v16bf frag_b_nmajor(const __bf16* p, int stride, int lane) {
  const __bf16* row = p + (lane & 15) * stride + ((lane >> 4) << 3);
  FragU f;
  f.u[0] = *(const uint4*)(row);
  f.u[1] = *(const uint4*)(row + 16);
  return f.v;
}

// --------------------------- f32 -> bf16 convert ---------------------------
__global__ __launch_bounds__(256)
void f32_to_bf16_kernel(const float* __restrict__ in, __bf16* __restrict__ out, int n) {
  int i = blockIdx.x * blockDim.x + threadIdx.x;
  int stride = gridDim.x * blockDim.x;
  for (; i < n; i += stride) out[i] = (__bf16)in[i];
}

// f32 [S][R][C] -> bf16 [S][C][R]  (weight pre-transpose to n-major)
__global__ __launch_bounds__(256)
void f32_to_bf16_T_kernel(const float* __restrict__ in, __bf16* __restrict__ out,
                          int S, int R, int C) {
  int i = blockIdx.x * blockDim.x + threadIdx.x;
  int total = S * R * C;
  int stride = gridDim.x * blockDim.x;
  for (; i < total; i += stride) {
    int s = i / (R * C);
    int rem = i - s * R * C;
    int r = rem / C, c = rem - r * C;
    out[(size_t)s * R * C + (size_t)c * R + r] = (__bf16)in[i];
  }
}

// ------------------------------- bf16 GEMM ---------------------------------
// C[M x Nc] = A[M x K] @ W + bias, where Wt is the n-major transpose [Nc][K].
// Block tile 128x64, BK=32, double-buffered async LDS staging.
// 8 waves each compute 32x32 (2x2 WMMA tiles).
// mode: 1=bias->f32, 2=bias+tanh->f32, 3=bias->bf16, 4=bias->bf16 transposed.
__global__ __launch_bounds__(256)
void gemm_bf16_kernel(const __bf16* __restrict__ A, const __bf16* __restrict__ Wt,
                      const float* __restrict__ bias, float* __restrict__ outF,
                      __bf16* __restrict__ outB, int M, int Nc, int K, int mode)
{
  __shared__ __align__(16) __bf16 As[2][128 * 32];   // row-major [m][k]
  __shared__ __align__(16) __bf16 Bt[2][64 * 32];    // n-major   [n][k]
  const int tid = threadIdx.x;
  const int lane = tid & 31, wid = tid >> 5;
  const int wm = wid >> 1, wn = wid & 1;             // 4x2 wave grid
  const int rowBase = blockIdx.x * 128;
  const int colBase = blockIdx.y * 64;

  // stage one K-tile (A: 512 x 16B chunks, B: 256 x 16B chunks)
  auto stage = [&](int buf, int kt) {
    int ch0 = tid;
    int r0 = ch0 >> 2, c40 = ch0 & 3;
    cp16((uint4*)As[buf] + ch0,
         (const uint4*)(A + (size_t)(rowBase + r0) * K + kt) + c40);
    int ch1 = tid + 256;
    int r1 = ch1 >> 2, c41 = ch1 & 3;
    cp16((uint4*)As[buf] + ch1,
         (const uint4*)(A + (size_t)(rowBase + r1) * K + kt) + c41);
    int n = tid >> 2, c4 = tid & 3;
    cp16((uint4*)Bt[buf] + tid,
         (const uint4*)(Wt + (size_t)(colBase + n) * K + kt) + c4);
  };

  v8f c00 = {0,0,0,0,0,0,0,0}, c01 = {0,0,0,0,0,0,0,0};
  v8f c10 = {0,0,0,0,0,0,0,0}, c11 = {0,0,0,0,0,0,0,0};

  const int nk = K / 32;
  stage(0, 0);
  for (int ki = 0; ki < nk; ++ki) {
    async_wait_all();
    __syncthreads();                         // buf[ki&1] fully staged
    if (ki + 1 < nk) stage((ki + 1) & 1, (ki + 1) * 32);  // overlap with WMMA

    const __bf16* as = As[ki & 1];
    const __bf16* bt = Bt[ki & 1];
    v16bf a0 = frag_a_rowmajor(as + (wm * 32) * 32, 32, lane);
    v16bf a1 = frag_a_rowmajor(as + (wm * 32 + 16) * 32, 32, lane);
    v16bf b0 = frag_b_nmajor(bt + (wn * 32) * 32, 32, lane);
    v16bf b1 = frag_b_nmajor(bt + (wn * 32 + 16) * 32, 32, lane);
    c00 = __builtin_amdgcn_wmma_f32_16x16x32_bf16(false, a0, false, b0, (short)0, c00, false, false);
    c01 = __builtin_amdgcn_wmma_f32_16x16x32_bf16(false, a0, false, b1, (short)0, c01, false, false);
    c10 = __builtin_amdgcn_wmma_f32_16x16x32_bf16(false, a1, false, b0, (short)0, c10, false, false);
    c11 = __builtin_amdgcn_wmma_f32_16x16x32_bf16(false, a1, false, b1, (short)0, c11, false, false);
    __syncthreads();                         // everyone done reading buf[ki&1]
  }

  // epilogue: C layout row = base + r + 8*(lane>>4), col = base + (lane&15)
  const int rofs = (lane >> 4) * 8;
  const int nl = lane & 15;
  v8f* cc[2][2] = {{&c00, &c01}, {&c10, &c11}};
  for (int i = 0; i < 2; ++i) {
    int mBase = rowBase + wm * 32 + i * 16;
    for (int j = 0; j < 2; ++j) {
      int nBase = colBase + wn * 32 + j * 16;
      int col = nBase + nl;
      float bcol = bias[col];
      for (int r = 0; r < 8; ++r) {
        int row = mBase + r + rofs;
        float v = (*cc[i][j])[r] + bcol;
        if (mode == 2) v = tanhf(v);
        if (outF) outF[(size_t)row * Nc + col] = v;
        if (outB) {
          if (mode == 4) outB[(size_t)col * M + row] = (__bf16)v;   // V^T
          else           outB[(size_t)row * Nc + col] = (__bf16)v;
        }
      }
    }
  }
}

// ------------------------------- LayerNorm ---------------------------------
__global__ __launch_bounds__(256)
void ln_rows_kernel(const float* __restrict__ X, const float* __restrict__ R,
                    const float* __restrict__ gamma, const float* __restrict__ beta,
                    float* __restrict__ outF, __bf16* __restrict__ outB, int Nrows)
{
  int lane = threadIdx.x & 31;
  int row = blockIdx.x * 8 + (threadIdx.x >> 5);
  if (row >= Nrows) return;
  const float* x = X + (size_t)row * DIM;
  const float* rr = R ? R + (size_t)row * DIM : nullptr;
  float v[8], s = 0.f, sq = 0.f;
  for (int j = 0; j < 8; ++j) {
    int col = j * 32 + lane;
    float t = x[col];
    if (rr) t += rr[col];
    v[j] = t; s += t; sq += t * t;
  }
  for (int o = 16; o > 0; o >>= 1) { s += __shfl_xor(s, o, 32); sq += __shfl_xor(sq, o, 32); }
  float mean = s * (1.f / DIM);
  float var  = sq * (1.f / DIM) - mean * mean;
  float rstd = rsqrtf(var + 1e-5f);
  for (int j = 0; j < 8; ++j) {
    int col = j * 32 + lane;
    float y = (v[j] - mean) * rstd * gamma[col] + beta[col];
    if (outF) outF[(size_t)row * DIM + col] = y;
    if (outB) outB[(size_t)row * DIM + col] = (__bf16)y;
  }
}

// ------------------------- fused block attention ---------------------------
// One workgroup per (batch, head). V comes pre-transposed: VT[DIM][NTOK].
__global__ __launch_bounds__(256)
void attn_block_kernel(const __bf16* __restrict__ Q, const __bf16* __restrict__ Kg,
                       const __bf16* __restrict__ VT, __bf16* __restrict__ O)
{
  __shared__ __align__(16) __bf16 Qs[128 * 32];   // [m][dh]
  __shared__ __align__(16) __bf16 Ks[128 * 32];   // [n][dh]
  __shared__ __align__(16) __bf16 Vt[32 * 128];   // [dh][m] (n-major for B frags)
  __shared__ __align__(16) __bf16 Ps[128 * 128];  // softmax probs, row-major
  int tid = threadIdx.x, lane = tid & 31, w = tid >> 5;
  int head = blockIdx.x & 7, batch = blockIdx.x >> 3;
  size_t base = (size_t)batch * 128 * DIM + (size_t)head * DHEAD;

  for (int ch = tid; ch < 512; ch += 256) {
    int r = ch >> 2, c4 = ch & 3;
    cp16((uint4*)Qs + ch, (const uint4*)(Q  + base + (size_t)r * DIM) + c4);
    cp16((uint4*)Ks + ch, (const uint4*)(Kg + base + (size_t)r * DIM) + c4);
  }
  for (int ch = tid; ch < 512; ch += 256) {
    int d = ch >> 4, c4 = ch & 15;
    cp16((uint4*)Vt + ch,
         (const uint4*)(VT + (size_t)(head * DHEAD + d) * NTOK + batch * 128) + c4);
  }
  async_wait_all();
  __syncthreads();

  // scores: wave w owns rows [w*16, w*16+16); K=32 == DH -> 1 WMMA per tile
  v8f sacc[8];
  v16bf aq = frag_a_rowmajor(Qs + (w * 16) * 32, 32, lane);
  for (int t = 0; t < 8; ++t) {
    v16bf bk = frag_b_nmajor(Ks + (t * 16) * 32, 32, lane);
    v8f z = {0,0,0,0,0,0,0,0};
    sacc[t] = __builtin_amdgcn_wmma_f32_16x16x32_bf16(false, aq, false, bk, (short)0, z, false, false);
  }

  // scale + softmax across 128 cols (8 tiles x 16 lanes of a half-wave)
  float inv[8];
  for (int r = 0; r < 8; ++r) {
    float mx = -3.4e38f;
    for (int t = 0; t < 8; ++t) { sacc[t][r] *= SCALE_QK; mx = fmaxf(mx, sacc[t][r]); }
    for (int o = 1; o < 16; o <<= 1) mx = fmaxf(mx, __shfl_xor(mx, o, 32));
    float sm = 0.f;
    for (int t = 0; t < 8; ++t) { float e = __expf(sacc[t][r] - mx); sacc[t][r] = e; sm += e; }
    for (int o = 1; o < 16; o <<= 1) sm += __shfl_xor(sm, o, 32);
    inv[r] = 1.f / sm;
  }
  const int rofs = (lane >> 4) * 8, nl = lane & 15;
  for (int t = 0; t < 8; ++t)
    for (int r = 0; r < 8; ++r)
      Ps[(w * 16 + r + rofs) * 128 + t * 16 + nl] = (__bf16)(sacc[t][r] * inv[r]);
  // wave only consumes its own P rows -> same-wave DS ordering suffices

  v8f oacc0 = {0,0,0,0,0,0,0,0}, oacc1 = {0,0,0,0,0,0,0,0};
  for (int kk = 0; kk < 4; ++kk) {
    v16bf ap  = frag_a_rowmajor(Ps + (w * 16) * 128 + kk * 32, 128, lane);
    v16bf bv0 = frag_b_nmajor(Vt + 0 * 128 + kk * 32, 128, lane);
    v16bf bv1 = frag_b_nmajor(Vt + 16 * 128 + kk * 32, 128, lane);
    oacc0 = __builtin_amdgcn_wmma_f32_16x16x32_bf16(false, ap, false, bv0, (short)0, oacc0, false, false);
    oacc1 = __builtin_amdgcn_wmma_f32_16x16x32_bf16(false, ap, false, bv1, (short)0, oacc1, false, false);
  }
  for (int r = 0; r < 8; ++r) {
    int token = batch * 128 + w * 16 + r + rofs;
    O[(size_t)token * DIM + head * DHEAD + 0  + nl] = (__bf16)oacc0[r];
    O[(size_t)token * DIM + head * DHEAD + 16 + nl] = (__bf16)oacc1[r];
  }
}

// ------------------------------- launcher ----------------------------------
extern "C" void kernel_launch(void* const* d_in, const int* in_sizes, int n_in,
                              void* d_out, int out_size, void* d_ws, size_t ws_size,
                              hipStream_t stream) {
  (void)in_sizes; (void)n_in; (void)out_size; (void)ws_size;
  const float* feat     = (const float*)d_in[0];
  // d_in[1] (ordering) unused: guaranteed repeat(arange(B), N/B) -> 128-blocks
  const float* lin_W    = (const float*)d_in[2];
  const float* lin_b    = (const float*)d_in[3];
  const float* lin_g    = (const float*)d_in[4];
  const float* lin_beta = (const float*)d_in[5];
  const float* Wq = (const float*)d_in[6];
  const float* Wk = (const float*)d_in[7];
  const float* Wv = (const float*)d_in[8];
  const float* bq = (const float*)d_in[9];
  const float* bk = (const float*)d_in[10];
  const float* bv = (const float*)d_in[11];
  const float* Wo = (const float*)d_in[12];
  const float* bo = (const float*)d_in[13];
  const float* lng = (const float*)d_in[14];
  const float* lnb = (const float*)d_in[15];
  float* out = (float*)d_out;

  char* ws = (char*)d_ws;
  size_t off = 0;
  auto wsalloc = [&](size_t bytes) -> void* {
    void* p = ws + off;
    off += (bytes + 255) & ~(size_t)255;
    return p;
  };
  const size_t ND = (size_t)NTOK * DIM;
  const size_t DD = (size_t)DIM * DIM;
  __bf16* b_feat = (__bf16*)wsalloc(ND * 2);
  __bf16* b_h    = (__bf16*)wsalloc(ND * 2);
  __bf16* b_q    = (__bf16*)wsalloc(ND * 2);
  __bf16* b_k    = (__bf16*)wsalloc(ND * 2);
  __bf16* b_vT   = (__bf16*)wsalloc(ND * 2);   // V stored transposed [D][N]
  __bf16* b_o    = (__bf16*)wsalloc(ND * 2);
  float*  f_h    = (float*)wsalloc(ND * 4);
  float*  f_tmp  = (float*)wsalloc(ND * 4);
  __bf16* b_linWt = (__bf16*)wsalloc(DD * 2);         // n-major transposes
  __bf16* b_WqT   = (__bf16*)wsalloc(2 * DD * 2);
  __bf16* b_WkT   = (__bf16*)wsalloc(2 * DD * 2);
  __bf16* b_WvT   = (__bf16*)wsalloc(2 * DD * 2);
  __bf16* b_WoT   = (__bf16*)wsalloc(2 * DD * 2);

  dim3 blk(256);
  f32_to_bf16_kernel<<<1024, blk, 0, stream>>>(feat, b_feat, (int)ND);
  f32_to_bf16_T_kernel<<<64,  blk, 0, stream>>>(lin_W, b_linWt, 1, DIM, DIM);
  f32_to_bf16_T_kernel<<<128, blk, 0, stream>>>(Wq, b_WqT, 2, DIM, DIM);
  f32_to_bf16_T_kernel<<<128, blk, 0, stream>>>(Wk, b_WkT, 2, DIM, DIM);
  f32_to_bf16_T_kernel<<<128, blk, 0, stream>>>(Wv, b_WvT, 2, DIM, DIM);
  f32_to_bf16_T_kernel<<<128, blk, 0, stream>>>(Wo, b_WoT, 2, DIM, DIM);

  dim3 ggrid(NTOK / 128, DIM / 64);

  // block 0 pre-layer: h = LN(tanh(feat @ lin_W + lin_b))
  gemm_bf16_kernel<<<ggrid, blk, 0, stream>>>(b_feat, b_linWt, lin_b, f_tmp, nullptr,
                                              NTOK, DIM, DIM, /*tanh*/2);
  ln_rows_kernel<<<NTOK / 8, blk, 0, stream>>>(f_tmp, nullptr, lin_g, lin_beta,
                                               f_h, b_h, NTOK);

  for (int i = 0; i < 2; ++i) {
    const __bf16* wqT = b_WqT + (size_t)i * DD;
    const __bf16* wkT = b_WkT + (size_t)i * DD;
    const __bf16* wvT = b_WvT + (size_t)i * DD;
    const __bf16* woT = b_WoT + (size_t)i * DD;
    gemm_bf16_kernel<<<ggrid, blk, 0, stream>>>(b_h, wqT, bq + i * DIM, nullptr, b_q,
                                                NTOK, DIM, DIM, 3);
    gemm_bf16_kernel<<<ggrid, blk, 0, stream>>>(b_h, wkT, bk + i * DIM, nullptr, b_k,
                                                NTOK, DIM, DIM, 3);
    gemm_bf16_kernel<<<ggrid, blk, 0, stream>>>(b_h, wvT, bv + i * DIM, nullptr, b_vT,
                                                NTOK, DIM, DIM, /*transposed*/4);
    attn_block_kernel<<<NBATCH * NHEAD, blk, 0, stream>>>(b_q, b_k, b_vT, b_o);
    gemm_bf16_kernel<<<ggrid, blk, 0, stream>>>(b_o, woT, bo + i * DIM, f_tmp, nullptr,
                                                NTOK, DIM, DIM, 1);
    if (i == 0) {
      ln_rows_kernel<<<NTOK / 8, blk, 0, stream>>>(f_h, f_tmp, lng, lnb,
                                                   f_h, b_h, NTOK);
    } else {
      ln_rows_kernel<<<NTOK / 8, blk, 0, stream>>>(f_h, f_tmp, lng + DIM, lnb + DIM,
                                                   out, nullptr, NTOK);
    }
  }
}
